// DMVFlow_62431644615319
// MI455X (gfx1250) — matrix operand order repaired and checked by reading, not compiled, not run on hardware
//
#include <hip/hip_runtime.h>
#include <hip/hip_bf16.h>
#include <math.h>

// Problem constants (from reference): B=256, T=256, D=768, K=128
#define M_TOTAL   65536      // B*T rows
#define DDIM      768        // contraction dim
#define NSTATE    128        // number of Gaussian states (GEMM N)
#define NCHUNK    24         // DDIM / 32
#define BROW_PAD  40         // LDS row stride in bf16 elems (64B data + 16B pad, from TDM pad)

// LDS layout (dynamic shared memory, byte offsets)
#define BUF_HI(p)   ((p) * 20480u)            // 128 rows * 80B
#define BUF_LO(p)   ((p) * 20480u + 10240u)
#define INV_OFF     40960u
#define SMEM_BYTES  (40960u + DDIM * 4u)      // 44032

typedef __attribute__((ext_vector_type(16))) __bf16 v16bf;
typedef __attribute__((ext_vector_type(8)))  __bf16 v8bf;
typedef __attribute__((ext_vector_type(8)))  float  v8f;
typedef __attribute__((ext_vector_type(4)))  unsigned int v4u;
typedef __attribute__((ext_vector_type(8)))  int v8i;
typedef __attribute__((ext_vector_type(4)))  int v4i;

// ---------------------------------------------------------------------------
// TDM: DMA one W chunk tile (128 rows x 64B) global -> LDS with HW row padding
// D# per CDNA5 ISA. data_size=4B units:
//   tensor_dim0=384, dim0_stride=384, tile_dim0=16 (64B), tile_dim1=128
//   pad_enable, pad_interval=3 (16 DWORDs = 64B), pad_amount=3 (4 DWORDs = 16B)
// -> LDS rows land at 80B stride (bank-conflict-free fragment reads).
// ---------------------------------------------------------------------------
__device__ inline void tdm_load_tile(const void* gaddr, unsigned lds_off) {
    unsigned long long ga = (unsigned long long)(uintptr_t)gaddr;
    v4u g0 = { 1u,                                    // count=1 (valid descriptor)
               lds_off,                               // lds_addr (bytes)
               (unsigned)(ga & 0xFFFFFFFFu),          // global_addr[31:0]
               (unsigned)((ga >> 32) & 0x01FFFFFFu)   // global_addr[56:32]
                   | (2u << 30) };                    // type=2 ("image")
    v8i g1 = { (int)((2u << 16)                       // data_size = 4B
                   | (1u << 20)                       // pad_enable
                   | (3u << 22)                       // pad_interval: 16 DWORDs
                   | (3u << 25)),                     // pad_amount: 4 DWORDs
               (int)(384u << 16),                     // tensor_dim0[15:0] = 384
               (int)(128u << 16),                     // dim0 hi=0 | tensor_dim1 = 128
               (int)(16u  << 16),                     // dim1 hi=0 | tile_dim0 = 16
               128,                                   // tile_dim1=128 | tile_dim2=0
               384,                                   // tensor_dim0_stride = 384
               0, 0 };                                // stride hi / dim1_stride = 0
    v4i z4 = {0, 0, 0, 0};
#if __clang_major__ >= 23
    v8i z8 = {0, 0, 0, 0, 0, 0, 0, 0};
    __builtin_amdgcn_tensor_load_to_lds(g0, g1, z4, z4, z8, 0);
#else
    __builtin_amdgcn_tensor_load_to_lds(g0, g1, z4, z4, 0);
#endif
}

// ---------------------------------------------------------------------------
// Block-wide sum reduction (256 threads)
// ---------------------------------------------------------------------------
__device__ inline float block_reduce_256(float v, float* sm) {
    int t = threadIdx.x;
    sm[t] = v;
    __syncthreads();
    #pragma unroll
    for (int s = 128; s > 0; s >>= 1) {
        if (t < s) sm[t] += sm[t + s];
        __syncthreads();
    }
    return sm[0];
}

// ---------------------------------------------------------------------------
// Prologue 1: log_norm = -0.5*(D*log(2*pi) + sum_d log(var[d]))   (1 block)
// ---------------------------------------------------------------------------
__global__ void dmv_lognorm_kernel(const float* __restrict__ var,
                                   float* __restrict__ lognorm_out) {
    __shared__ float sm[256];
    float p = 0.f;
    for (int d = threadIdx.x; d < DDIM; d += 256) p += logf(var[d]);
    float tot = block_reduce_256(p, sm);
    if (threadIdx.x == 0) {
        lognorm_out[0] = -0.5f * ((float)DDIM * logf(6.283185307179586f) + tot);
    }
}

// ---------------------------------------------------------------------------
// Prologue 2: per-state k (128 blocks):
//   W[k][d] = means[k][d] / var[d]  split into bf16 hi/lo halves
//   c[k]    = log_norm - 0.5 * sum_d means[k][d]^2 / var[d]
// ---------------------------------------------------------------------------
__global__ void dmv_prep_kernel(const float* __restrict__ means,
                                const float* __restrict__ var,
                                const float* __restrict__ lognorm,
                                __bf16* __restrict__ Whi,
                                __bf16* __restrict__ Wlo,
                                float* __restrict__ cks) {
    __shared__ float sm[256];
    int k = blockIdx.x;
    float m2p = 0.f;
    for (int d = threadIdx.x; d < DDIM; d += 256) {
        float iv = 1.0f / var[d];
        float mv = means[(size_t)k * DDIM + d];
        float w  = mv * iv;
        __bf16 h = (__bf16)w;
        Whi[(size_t)k * DDIM + d] = h;
        Wlo[(size_t)k * DDIM + d] = (__bf16)(w - (float)h);
        m2p += mv * mv * iv;
    }
    float m2 = block_reduce_256(m2p, sm);
    if (threadIdx.x == 0) cks[k] = lognorm[0] - 0.5f * m2;
}

// ---------------------------------------------------------------------------
// Main kernel: 512 blocks x 256 threads (8 waves). Wave w of block b computes
// output rows [ (b*8+w)*16 , +16 ) x all 128 states via 8 WMMA tiles.
// Split-bf16 fp32 emulation: acc += Ahi*Bhi + Ahi*Blo + Alo*Bhi.
// W chunks staged by the Tensor Data Mover, double-buffered in LDS.
// ---------------------------------------------------------------------------
__global__ void __launch_bounds__(256)
dmv_density_kernel(const float* __restrict__ s,
                   const float* __restrict__ var,
                   const __bf16* __restrict__ Whi,
                   const __bf16* __restrict__ Wlo,
                   const float* __restrict__ cks,
                   float* __restrict__ out) {
    extern __shared__ char smem[];
    float* sInv = (float*)(smem + INV_OFF);

    const int tid  = threadIdx.x;
    const int lane = tid & 31;
    const int wave = tid >> 5;

    // one-time: inverse variance into LDS (visible after first barrier)
    for (int d = tid; d < DDIM; d += 256) sInv[d] = 1.0f / var[d];

    // TDM prologue: wave 0 issues chunk 0 and chunk 1 DMAs (in-order per wave)
    if (wave == 0) {
        tdm_load_tile(Whi + 0,  BUF_HI(0));
        tdm_load_tile(Wlo + 0,  BUF_LO(0));
        tdm_load_tile(Whi + 32, BUF_HI(1));
        tdm_load_tile(Wlo + 32, BUF_LO(1));
    }

    const int m0  = (blockIdx.x * 8 + wave) * 16;       // first output row of this wave
    const int row = m0 + (lane & 15);                   // A row handled by this lane
    const int g8  = (lane >> 4) << 3;                   // 0 or 8: lane-half K offset
    const float* sp = s + (size_t)row * DDIM;

    v8f acc[8] = {};                                    // 8 N-tiles of 16x16 f32
    float s2p = 0.f;                                    // partial s^2/var for this lane's half

    for (int c = 0; c < NCHUNK; ++c) {
        const int d0 = c * 32;
        // wave 0: wait until chunk c's two TDM ops landed (<=2 outstanding = chunk c+1)
        if (wave == 0) {
            if (c + 1 < NCHUNK) __builtin_amdgcn_s_wait_tensorcnt(2);
            else                __builtin_amdgcn_s_wait_tensorcnt(0);
        }
        __syncthreads();                                // chunk c visible to all waves

        const __bf16* sBhi = (const __bf16*)(smem + BUF_HI(c & 1));
        const __bf16* sBlo = (const __bf16*)(smem + BUF_LO(c & 1));

        // ---- A fragment: 16-bit A 16x32 layout (ISA 7.12.2) ----
        // lane l holds row l%16, K = d0 + 8*(l/16) + [0..8) and +16
        const int kb = d0 + g8;
        float4 x0 = *(const float4*)(sp + kb);
        float4 x1 = *(const float4*)(sp + kb + 4);
        float4 x2 = *(const float4*)(sp + kb + 16);
        float4 x3 = *(const float4*)(sp + kb + 20);
        if (d0 + 32 < DDIM) {
            __builtin_prefetch(sp + kb + 32, 0, 0);     // global_prefetch_b8: next chunk of s
            __builtin_prefetch(sp + kb + 48, 0, 0);
        }
        float a[16] = { x0.x, x0.y, x0.z, x0.w, x1.x, x1.y, x1.z, x1.w,
                        x2.x, x2.y, x2.z, x2.w, x3.x, x3.y, x3.z, x3.w };
        v16bf ahi, alo;
        #pragma unroll
        for (int i = 0; i < 16; ++i) {
            float x = a[i];
            __bf16 h = (__bf16)x;
            ahi[i] = h;
            alo[i] = (__bf16)(x - (float)h);
            int dd = kb + i + ((i >= 8) ? 8 : 0);       // actual d index of element i
            s2p += x * x * sInv[dd];
        }

        // ---- 8 N-tiles: B fragment (B 32x16: lane l = col l%16, K = 8*(l/16) + [0..8) and +16) ----
        #pragma unroll
        for (int j = 0; j < 8; ++j) {
            const int boff = (16 * j + (lane & 15)) * BROW_PAD + ((lane >> 4) << 3);
            v8bf h0 = *(const v8bf*)(sBhi + boff);
            v8bf h1 = *(const v8bf*)(sBhi + boff + 16);
            v8bf l0 = *(const v8bf*)(sBlo + boff);
            v8bf l1 = *(const v8bf*)(sBlo + boff + 16);
            v16bf bhi = __builtin_shufflevector(h0, h1, 0,1,2,3,4,5,6,7,8,9,10,11,12,13,14,15);
            v16bf blo = __builtin_shufflevector(l0, l1, 0,1,2,3,4,5,6,7,8,9,10,11,12,13,14,15);
            acc[j] = __builtin_amdgcn_wmma_f32_16x16x32_bf16(false, ahi, false, bhi,
                                                             (short)0, acc[j], false, false);
            acc[j] = __builtin_amdgcn_wmma_f32_16x16x32_bf16(false, ahi, false, blo,
                                                             (short)0, acc[j], false, false);
            acc[j] = __builtin_amdgcn_wmma_f32_16x16x32_bf16(false, alo, false, bhi,
                                                             (short)0, acc[j], false, false);
        }

        __syncthreads();                                // all waves done with buf (c&1)
        if (wave == 0 && c + 2 < NCHUNK) {              // refill it with chunk c+2
            tdm_load_tile(Whi + (size_t)(d0 + 64), BUF_HI(c & 1));
            tdm_load_tile(Wlo + (size_t)(d0 + 64), BUF_LO(c & 1));
        }
    }

    // ---- epilogue: density = c[k] + cross - 0.5*s2 ----
    float s2f = s2p + __shfl_xor(s2p, 16, 32);          // combine lane-half K partials
    const int mbase = (lane >> 4) << 3;                 // C layout: VGPR r -> M = r + 8*(l/16)
    #pragma unroll
    for (int j = 0; j < 8; ++j) {
        const int n = 16 * j + (lane & 15);
        const float ck = cks[n];
        #pragma unroll
        for (int r = 0; r < 8; ++r) {
            float s2r = __shfl(s2f, mbase + r, 32);     // s2 of row (mbase+r) lives in that lane
            out[(size_t)(m0 + mbase + r) * NSTATE + n] = ck + acc[j][r] - 0.5f * s2r;
        }
    }
}

// ---------------------------------------------------------------------------
// Launch
// ---------------------------------------------------------------------------
extern "C" void kernel_launch(void* const* d_in, const int* in_sizes, int n_in,
                              void* d_out, int out_size, void* d_ws, size_t ws_size,
                              hipStream_t stream) {
    const float* s     = (const float*)d_in[0];   // (B,T,D)
    const float* means = (const float*)d_in[1];   // (K,D)
    const float* var   = (const float*)d_in[2];   // (D,)
    float* out = (float*)d_out;                   // (B,T,K) f32

    // workspace layout
    float*  cks     = (float*)d_ws;                          // 128 floats
    float*  lognorm = cks + NSTATE;                          // 1 float
    __bf16* Whi     = (__bf16*)((char*)d_ws + 1024);         // 128*768 bf16
    __bf16* Wlo     = Whi + (size_t)NSTATE * DDIM;           // 128*768 bf16

    dmv_lognorm_kernel<<<1, 256, 0, stream>>>(var, lognorm);
    dmv_prep_kernel<<<NSTATE, 256, 0, stream>>>(means, var, lognorm, Whi, Wlo, cks);
    dmv_density_kernel<<<M_TOTAL / (16 * 8), 256, SMEM_BYTES, stream>>>(s, var, Whi, Wlo, cks, out);
}
